// Attention_45758581571928
// MI455X (gfx1250) — compile-verified, matrix operand order checked by reading, not run on hardware
//
#include <hip/hip_runtime.h>
#include <hip/hip_bf16.h>

// ---------------------------------------------------------------------------
// Attention block for MI455X (gfx1250): bf16 WMMA GEMMs + flash attention,
// with GLOBAL_LOAD_ASYNC_TO_LDS_B128 staging (ASYNCcnt) and double buffering.
// ---------------------------------------------------------------------------

#define B_ 2
#define T_ 2048
#define C_ 2048
#define NH_ 16
#define NKV_ 4
#define HD_ 128
#define BT_ (B_ * T_)

typedef __bf16 bf16;
typedef bf16  v16bf __attribute__((ext_vector_type(16)));
typedef bf16  v8bf  __attribute__((ext_vector_type(8)));
typedef float v8f   __attribute__((ext_vector_type(8)));

union Frag16 { v16bf v; v8bf h[2]; };

// WMMA 16-bit A/B fragment from a row-major LDS row pointer.
// Lane L: row = L&15, kbase = (L>>4)*8 holds K = kbase..+7 and kbase+16..+23.
__device__ inline v16bf load_frag(const bf16* rowp, int kbase) {
    Frag16 f;
    f.h[0] = *(const v8bf*)(rowp + kbase);
    f.h[1] = *(const v8bf*)(rowp + kbase + 16);
    return f.v;
}

__device__ inline v8f wmma_bf16(v16bf a, v16bf b, v8f c) {
    return __builtin_amdgcn_wmma_f32_16x16x32_bf16(
        false, a, false, b, (short)0, c, false, false);
}

// CDNA5 async memory->LDS copy (tracked by ASYNCcnt). Per-lane 16B.
__device__ inline void async_ld_b128(unsigned lds_off, const bf16* g) {
    asm volatile("global_load_async_to_lds_b128 %0, %1, off"
                 :: "v"(lds_off), "v"((unsigned long long)g)
                 : "memory");
}
__device__ inline void wait_async0() {
    asm volatile("s_wait_asynccnt 0" ::: "memory");
}
__device__ inline unsigned lds_off_of(const bf16* p) {
    // generic pointers to LDS carry the LDS byte offset in the low 32 bits
    return (unsigned)(unsigned long long)p;
}

// ---------------------------------------------------------------------------
// Elementwise converts / transposes
// ---------------------------------------------------------------------------
__global__ void k_f32_to_bf16(const float* __restrict__ in,
                              bf16* __restrict__ out, long long n) {
    long long i = (long long)blockIdx.x * blockDim.x + threadIdx.x;
    if (i < n) out[i] = (bf16)in[i];
}

// in: [K][N] f32 row-major  ->  out: [N][K] bf16 row-major (B-transposed)
__global__ void k_transpose_bf16(const float* __restrict__ in,
                                 bf16* __restrict__ out, int K, int N) {
    long long i = (long long)blockIdx.x * blockDim.x + threadIdx.x;
    long long total = (long long)K * N;
    if (i >= total) return;
    int k = (int)(i % K);
    int n = (int)(i / K);
    out[(long long)n * K + k] = (bf16)in[(long long)k * N + n];
}

// RoPE + relayout: q f32 [B,T,H,D] token-major -> qh bf16 [B,H,T,D]
__global__ void k_rope_heads(const float* __restrict__ q,
                             const float* __restrict__ cache, // [T, HD/2, 2]
                             bf16* __restrict__ qh, int H) {
    long long i = (long long)blockIdx.x * blockDim.x + threadIdx.x;
    long long total = (long long)B_ * T_ * H * (HD_ / 2);
    if (i >= total) return;
    int d2 = (int)(i % (HD_ / 2));
    long long r = i / (HD_ / 2);
    int h = (int)(r % H); r /= H;
    int t = (int)(r % T_);
    int b = (int)(r / T_);
    float c = cache[((long long)t * (HD_ / 2) + d2) * 2 + 0];
    float s = cache[((long long)t * (HD_ / 2) + d2) * 2 + 1];
    long long ib = (((long long)(b * T_ + t)) * H + h) * HD_ + 2 * d2;
    float x0 = q[ib], x1 = q[ib + 1];
    long long ob = (((long long)(b * H + h)) * T_ + t) * HD_ + 2 * d2;
    qh[ob]     = (bf16)(x0 * c - x1 * s);
    qh[ob + 1] = (bf16)(x0 * s + x1 * c);
}

// v f32 [B,T,NKV,HD] -> vt bf16 [B,NKV,HD,T]  (so V is a ready-made B^T)
__global__ void k_v_transpose(const float* __restrict__ v,
                              bf16* __restrict__ vt) {
    long long i = (long long)blockIdx.x * blockDim.x + threadIdx.x;
    long long total = (long long)B_ * NKV_ * HD_ * T_;
    if (i >= total) return;
    int t = (int)(i % T_);
    long long r = i / T_;
    int d = (int)(r % HD_); r /= HD_;
    int h = (int)(r % NKV_);
    int b = (int)(r / NKV_);
    vt[(((long long)(b * NKV_ + h)) * HD_ + d) * T_ + t] =
        (bf16)v[(((long long)(b * T_ + t)) * NKV_ + h) * HD_ + d];
}

// ---------------------------------------------------------------------------
// bf16 GEMM: C[M,N] = A[M,K] * Bt[N,K]^T. Block tile 128x128, BK=64,
// double-buffered LDS filled by async global->LDS loads. 8 waves, each
// computing a 32x64 sub-tile (2x4 WMMA accumulators, 8 WMMA / k-step).
// ---------------------------------------------------------------------------
__global__ __launch_bounds__(256)
void k_gemm_bf16(const bf16* __restrict__ A, const bf16* __restrict__ Bt,
                 float* __restrict__ C, int M, int N, int K) {
    __shared__ __align__(16) bf16 sA[2][128 * 64];
    __shared__ __align__(16) bf16 sB[2][128 * 64];

    const int tid  = threadIdx.x;
    const int lane = tid & 31;
    const int wave = tid >> 5;
    const int wm = wave & 3;    // 0..3 : 32-row band
    const int wn = wave >> 2;   // 0..1 : 64-col band
    const int m0 = blockIdx.y * 128;
    const int n0 = blockIdx.x * 128;

    const int lm    = lane & 15;
    const int kbase = (lane >> 4) * 8;
    const int half  = lane >> 4;

    v8f acc[2][4];
#pragma unroll
    for (int ti = 0; ti < 2; ++ti)
#pragma unroll
        for (int tj = 0; tj < 4; ++tj)
#pragma unroll
            for (int e = 0; e < 8; ++e) acc[ti][tj][e] = 0.0f;

    // Stage issue: 128x64 bf16 tile per operand = 1024 16B chunks;
    // 256 threads x 4 chunks each (64B contiguous per thread -> coalesced).
    auto issue_stage = [&](int buf, int kt) {
#pragma unroll
        for (int q = 0; q < 4; ++q) {
            int c   = tid * 4 + q;
            int row = c >> 3, col8 = (c & 7) * 8;
            async_ld_b128(lds_off_of(&sA[buf][row * 64 + col8]),
                          A + (long long)(m0 + row) * K + kt + col8);
        }
#pragma unroll
        for (int q = 0; q < 4; ++q) {
            int c   = tid * 4 + q;
            int row = c >> 3, col8 = (c & 7) * 8;
            async_ld_b128(lds_off_of(&sB[buf][row * 64 + col8]),
                          Bt + (long long)(n0 + row) * K + kt + col8);
        }
    };

    const int nstages = K / 64;
    issue_stage(0, 0);
    wait_async0();
    __syncthreads();

    int buf = 0;
    for (int s = 0; s < nstages; ++s) {
        if (s + 1 < nstages) issue_stage(buf ^ 1, (s + 1) * 64);

        const bf16* a_base = &sA[buf][0];
        const bf16* b_base = &sB[buf][0];
#pragma unroll
        for (int ks = 0; ks < 2; ++ks) {
            v16bf af[2], bfr[4];
#pragma unroll
            for (int ti = 0; ti < 2; ++ti)
                af[ti] = load_frag(a_base + (wm * 32 + ti * 16 + lm) * 64 + ks * 32,
                                   kbase);
#pragma unroll
            for (int tj = 0; tj < 4; ++tj)
                bfr[tj] = load_frag(b_base + (wn * 64 + tj * 16 + lm) * 64 + ks * 32,
                                    kbase);
#pragma unroll
            for (int ti = 0; ti < 2; ++ti)
#pragma unroll
                for (int tj = 0; tj < 4; ++tj)
                    acc[ti][tj] = wmma_bf16(af[ti], bfr[tj], acc[ti][tj]);
        }

        wait_async0();       // next-stage tiles resident in LDS
        __syncthreads();     // and everyone done reading current buffer
        buf ^= 1;
    }

    // C/D layout: lane l, vgpr r -> M = r + 8*(l>=16), N = l&15
#pragma unroll
    for (int ti = 0; ti < 2; ++ti)
#pragma unroll
        for (int tj = 0; tj < 4; ++tj)
#pragma unroll
            for (int r = 0; r < 8; ++r) {
                int m = m0 + wm * 32 + ti * 16 + r + 8 * half;
                int n = n0 + wn * 64 + tj * 16 + lm;
                C[(long long)m * N + n] = acc[ti][tj][r];
            }
}

// ---------------------------------------------------------------------------
// Flash attention (causal, GQA). Grid: (B*NH, T/64). Block: 128 thr = 4 waves.
// Each wave owns 16 query rows; key tiles of 32; online softmax.
// qh: [B,NH,T,HD] bf16   kh: [B,NKV,T,HD] bf16   vt: [B,NKV,HD,T] bf16
// y : [B,T,NH,HD] f32
// ---------------------------------------------------------------------------
__global__ __launch_bounds__(128)
void k_flash_attn(const bf16* __restrict__ qh, const bf16* __restrict__ kh,
                  const bf16* __restrict__ vt, float* __restrict__ y) {
    __shared__ __align__(16) bf16 Qs[64 * 128];
    __shared__ __align__(16) bf16 Ks[32 * 128];
    __shared__ __align__(16) bf16 Vts[128 * 32];
    __shared__ __align__(16) bf16 Ps[4][16 * 32];

    const int bh  = blockIdx.x;
    const int b   = bh / NH_, h = bh % NH_;
    const int kvh = h / (NH_ / NKV_);
    const int q0  = blockIdx.y * 64;
    const int tid = threadIdx.x, lane = tid & 31, wave = tid >> 5;
    const int lm = lane & 15, kbase = (lane >> 4) * 8, half = lane >> 4;

    const bf16* Qg  = qh + (((long long)(b * NH_ + h)) * T_ + q0) * HD_;
    const bf16* Kg  = kh + (((long long)(b * NKV_ + kvh)) * T_) * HD_;
    const bf16* Vtg = vt + (((long long)(b * NKV_ + kvh)) * HD_) * (long long)T_;

    // stage Q tile 64x128 (16 KB): 1024 chunks, 8 per thread, async
#pragma unroll
    for (int q = 0; q < 8; ++q) {
        int c = tid * 8 + q;            // 128B contiguous per thread
        int row = c >> 4, col8 = (c & 15) * 8;
        async_ld_b128(lds_off_of(&Qs[row * 128 + col8]),
                      Qg + (long long)row * HD_ + col8);
    }

    v8f o[8];
#pragma unroll
    for (int t = 0; t < 8; ++t)
#pragma unroll
        for (int e = 0; e < 8; ++e) o[t][e] = 0.0f;

    float mrun[8], lrun[8];
#pragma unroll
    for (int r = 0; r < 8; ++r) { mrun[r] = -1e30f; lrun[r] = 0.0f; }

    const float scale = 0.08838834764831845f; // 1/sqrt(128)
    const int nkb = q0 / 32 + 2;              // causal key blocks

    for (int kb = 0; kb < nkb; ++kb) {
        __syncthreads();   // everyone done reading previous K/V tiles
        // K tile 32x128 = 512 chunks, 4 per thread
#pragma unroll
        for (int q = 0; q < 4; ++q) {
            int c = tid * 4 + q;
            int row = c >> 4, col8 = (c & 15) * 8;
            async_ld_b128(lds_off_of(&Ks[row * 128 + col8]),
                          Kg + ((long long)kb * 32 + row) * HD_ + col8);
        }
        // Vt tile 128x32 = 512 chunks, 4 per thread
#pragma unroll
        for (int q = 0; q < 4; ++q) {
            int c = tid * 4 + q;
            int d = c >> 2, j8 = (c & 3) * 8;
            async_ld_b128(lds_off_of(&Vts[d * 32 + j8]),
                          Vtg + (long long)d * T_ + kb * 32 + j8);
        }
        wait_async0();
        __syncthreads();

        // S = Q K^T : 16 queries x 32 keys, accumulate over d=128
        v8f s[2];
#pragma unroll
        for (int t = 0; t < 2; ++t)
#pragma unroll
            for (int e = 0; e < 8; ++e) s[t][e] = 0.0f;
#pragma unroll
        for (int ks = 0; ks < 4; ++ks) {
            v16bf a  = load_frag(&Qs[(wave * 16 + lm) * 128 + ks * 32], kbase);
            v16bf b0 = load_frag(&Ks[(0  + lm) * 128 + ks * 32], kbase);
            v16bf b1 = load_frag(&Ks[(16 + lm) * 128 + ks * 32], kbase);
            s[0] = wmma_bf16(a, b0, s[0]);
            s[1] = wmma_bf16(a, b1, s[1]);
        }

        // scale + causal mask + online softmax (rows r of this lane's half)
        float fac[8];
#pragma unroll
        for (int r = 0; r < 8; ++r) {
            int qi = q0 + wave * 16 + r + 8 * half;
            float vmax = -1e30f;
#pragma unroll
            for (int t = 0; t < 2; ++t) {
                int kj = kb * 32 + t * 16 + lm;
                float sv = s[t][r] * scale;
                sv = (kj <= qi) ? sv : -1e30f;
                s[t][r] = sv;
                vmax = fmaxf(vmax, sv);
            }
#pragma unroll
            for (int msk = 1; msk < 16; msk <<= 1)
                vmax = fmaxf(vmax, __shfl_xor(vmax, msk, 32));
            float mnew = fmaxf(mrun[r], vmax);
            float f = __expf(mrun[r] - mnew);
            float rsum = 0.0f;
#pragma unroll
            for (int t = 0; t < 2; ++t) {
                float p = __expf(s[t][r] - mnew);
                s[t][r] = p;
                rsum += p;
            }
#pragma unroll
            for (int msk = 1; msk < 16; msk <<= 1)
                rsum += __shfl_xor(rsum, msk, 32);
            lrun[r] = lrun[r] * f + rsum;
            mrun[r] = mnew;
            fac[r] = f;
        }
#pragma unroll
        for (int t = 0; t < 8; ++t)
#pragma unroll
            for (int r = 0; r < 8; ++r) o[t][r] *= fac[r];

        // P (C-layout f32) -> per-wave LDS (row-major bf16) to re-fragment
        bf16* Pw = Ps[wave];
#pragma unroll
        for (int t = 0; t < 2; ++t)
#pragma unroll
            for (int r = 0; r < 8; ++r) {
                int m = r + 8 * half;
                int j = t * 16 + lm;
                Pw[m * 32 + j] = (bf16)s[t][r];
            }
        // O += P * V   (A = P 16x32, B^T = Vts rows d)
        v16bf pa = load_frag(&Pw[lm * 32], kbase);
#pragma unroll
        for (int t = 0; t < 8; ++t) {
            v16bf vb = load_frag(&Vts[(t * 16 + lm) * 32], kbase);
            o[t] = wmma_bf16(pa, vb, o[t]);
        }
    }

    // normalize + store y[B,T,NH,HD]
    float inv[8];
#pragma unroll
    for (int r = 0; r < 8; ++r) inv[r] = 1.0f / lrun[r];
#pragma unroll
    for (int t = 0; t < 8; ++t)
#pragma unroll
        for (int r = 0; r < 8; ++r) {
            int ti = q0 + wave * 16 + r + 8 * half;
            int d  = t * 16 + lm;
            y[(((long long)b * T_ + ti) * NH_ + h) * HD_ + d] = o[t][r] * inv[r];
        }
}

// ---------------------------------------------------------------------------
// Launch
// ---------------------------------------------------------------------------
extern "C" void kernel_launch(void* const* d_in, const int* in_sizes, int n_in,
                              void* d_out, int out_size, void* d_ws, size_t ws_size,
                              hipStream_t stream) {
    const float* x     = (const float*)d_in[0];
    const float* cache = (const float*)d_in[1];
    const float* Wq    = (const float*)d_in[2];
    const float* Wk    = (const float*)d_in[3];
    const float* Wv    = (const float*)d_in[4];
    const float* Wo    = (const float*)d_in[5];
    float* out = (float*)d_out;

    char* ws = (char*)d_ws;
    size_t off = 0;
    auto alloc = [&](size_t bytes) {
        void* p = ws + off;
        off = (off + bytes + 255) & ~(size_t)255;
        return p;
    };

    bf16* xbf = (bf16*)alloc((size_t)BT_ * C_ * 2);
    bf16* wqT = (bf16*)alloc((size_t)C_ * C_ * 2);
    bf16* wkT = (bf16*)alloc((size_t)(NKV_ * HD_) * C_ * 2);
    bf16* wvT = (bf16*)alloc((size_t)(NKV_ * HD_) * C_ * 2);
    bf16* woT = (bf16*)alloc((size_t)C_ * C_ * 2);
    float* qf = (float*)alloc((size_t)BT_ * C_ * 4);
    float* kf = (float*)alloc((size_t)BT_ * NKV_ * HD_ * 4);
    float* vf = (float*)alloc((size_t)BT_ * NKV_ * HD_ * 4);
    bf16* qhh = (bf16*)alloc((size_t)B_ * NH_ * T_ * HD_ * 2);
    bf16* khh = (bf16*)alloc((size_t)B_ * NKV_ * T_ * HD_ * 2);
    bf16* vtt = (bf16*)alloc((size_t)B_ * NKV_ * HD_ * T_ * 2);
    float* ya = (float*)alloc((size_t)BT_ * C_ * 4);
    bf16* ybf = (bf16*)alloc((size_t)BT_ * C_ * 2);

    const int TPB = 256;
    auto blocks = [&](long long n) { return (unsigned)((n + TPB - 1) / TPB); };

    // 1) converts / transposes
    k_f32_to_bf16<<<blocks((long long)BT_ * C_), TPB, 0, stream>>>(
        x, xbf, (long long)BT_ * C_);
    k_transpose_bf16<<<blocks((long long)C_ * C_), TPB, 0, stream>>>(
        Wq, wqT, C_, C_);
    k_transpose_bf16<<<blocks((long long)C_ * NKV_ * HD_), TPB, 0, stream>>>(
        Wk, wkT, C_, NKV_ * HD_);
    k_transpose_bf16<<<blocks((long long)C_ * NKV_ * HD_), TPB, 0, stream>>>(
        Wv, wvT, C_, NKV_ * HD_);
    k_transpose_bf16<<<blocks((long long)C_ * C_), TPB, 0, stream>>>(
        Wo, woT, C_, C_);

    // 2) QKV projections
    k_gemm_bf16<<<dim3(C_ / 128, BT_ / 128), 256, 0, stream>>>(
        xbf, wqT, qf, BT_, C_, C_);
    k_gemm_bf16<<<dim3((NKV_ * HD_) / 128, BT_ / 128), 256, 0, stream>>>(
        xbf, wkT, kf, BT_, NKV_ * HD_, C_);
    k_gemm_bf16<<<dim3((NKV_ * HD_) / 128, BT_ / 128), 256, 0, stream>>>(
        xbf, wvT, vf, BT_, NKV_ * HD_, C_);

    // 3) RoPE + head-major relayout
    k_rope_heads<<<blocks((long long)B_ * T_ * NH_ * (HD_ / 2)), TPB, 0, stream>>>(
        qf, cache, qhh, NH_);
    k_rope_heads<<<blocks((long long)B_ * T_ * NKV_ * (HD_ / 2)), TPB, 0, stream>>>(
        kf, cache, khh, NKV_);
    k_v_transpose<<<blocks((long long)B_ * NKV_ * HD_ * T_), TPB, 0, stream>>>(
        vf, vtt);

    // 4) flash attention
    k_flash_attn<<<dim3(B_ * NH_, T_ / 64), 128, 0, stream>>>(qhh, khh, vtt, ya);

    // 5) output projection
    k_f32_to_bf16<<<blocks((long long)BT_ * C_), TPB, 0, stream>>>(
        ya, ybf, (long long)BT_ * C_);
    k_gemm_bf16<<<dim3(C_ / 128, BT_ / 128), 256, 0, stream>>>(
        ybf, woT, out, BT_, C_, C_);
}
